// TopKGating_81870666596847
// MI455X (gfx1250) — matrix operand order repaired and checked by reading, not compiled, not run on hardware
//
#include <hip/hip_runtime.h>
#include <hip/hip_bf16.h>

#define NUM_TOKENS 16384
#define INPUT_DIM  4096
#define NUM_EXPERTS 64
#define TOP_K 2

#define TOK_TILE 128                   // tokens per block
#define KC 32                          // K per bf16 WMMA
#define NCHUNK (INPUT_DIM / KC)        // 128 K-chunks
#define NTILE  (NUM_EXPERTS / 16)      // 4 expert tiles of N=16
#define XS_STRIDE 40                   // 32 halves + pad, keeps 16B align + bank spread

typedef __attribute__((ext_vector_type(16))) __bf16 v16bf;
typedef __attribute__((ext_vector_type(8)))  __bf16 v8bf;
typedef __attribute__((ext_vector_type(8)))  float  v8f;

// ---------------------------------------------------------------------------
// Kernel 1: pack W (fp32, [64 experts][4096 dims]) into bf16 B-fragments laid
// out for V_WMMA_F32_16X16X32_BF16's B operand in contiguous K order (matches A):
//   B is 32x16 (KxN); lane l holds N = l&15, K half = (l>>4)*16, and the 16
//   halves in the lane are consecutive K pairs (half h <-> K = (l>>4)*16 + h).
// Layout: Bpack[chunk][tile][lane][h], so the GEMM kernel loads one
// contiguous 32B vector per lane (two global_load_b128, fully coalesced).
// ---------------------------------------------------------------------------
__global__ void topk_gating_pack_w(const float* __restrict__ W,
                                   __bf16* __restrict__ Bp) {
    int tid  = blockIdx.x * blockDim.x + threadIdx.x;   // 0 .. 262143
    int h    =  tid        & 15;
    int lane = (tid >> 4)  & 31;
    int tile = (tid >> 9)  & 3;
    int c    =  tid >> 11;
    int k = c * KC + ((lane >> 4) << 4) + h;            // global dim index
    int e = tile * 16 + (lane & 15);                    // expert index
    Bp[tid] = (__bf16)W[(size_t)e * INPUT_DIM + k];
}

// ---------------------------------------------------------------------------
// Kernel 2: GEMM (bf16 WMMA, fp32 accum) + softmax + top-2 per token.
// 256 threads = 8 waves; wave w owns tokens [w*16, w*16+16) of the block's
// 128-token tile and all 64 experts (4 accumulator tiles).
// Pipelining: B-fragment global loads (L2-resident) are issued at the top of
// each chunk iteration so their latency hides under the x staging + barrier;
// the 4 WMMAs then issue back-to-back on independent accumulators.
// ---------------------------------------------------------------------------
__global__ void __launch_bounds__(256)
topk_gating_gemm(const float* __restrict__ x,
                 const __bf16* __restrict__ Bpack,
                 const float* __restrict__ bias,
                 float* __restrict__ out) {
    __shared__ __align__(16) __bf16 xs[2][TOK_TILE][XS_STRIDE]; // staged x (bf16)
    __shared__ float sc[TOK_TILE][NUM_EXPERTS + 1];             // score tile
    __shared__ float bsh[NUM_EXPERTS];

    const int tid  = threadIdx.x;
    const int wave = tid >> 5;
    const int lane = tid & 31;
    const int base = blockIdx.x * TOK_TILE;

    if (tid < NUM_EXPERTS) bsh[tid] = bias[tid];

    const v8f vzero = {0.f, 0.f, 0.f, 0.f, 0.f, 0.f, 0.f, 0.f};
    v8f acc[NTILE];
#pragma unroll
    for (int t = 0; t < NTILE; ++t) acc[t] = vzero;

    // A-fragment addressing (16-bit A 16x32, contiguous K order):
    // lane l: row M = l&15; lanes 0-15 hold K in {0..15}, lanes 16-31
    // hold K in {16..31} (matches B pack).  -> two contiguous 16B LDS reads per lane.
    const int m    = lane & 15;
    const int row  = wave * 16 + m;
    const int koff = (lane >> 4) * 16;
    const v16bf* Bp = (const v16bf*)Bpack;

    for (int c = 0; c < NCHUNK; ++c) {
        const int buf = c & 1;
        const int kc  = c * KC;

        // Issue all 4 B-fragment loads first: independent of LDS, latency
        // overlaps the x staging below and the barrier.
        v16bf bfrag[NTILE];
#pragma unroll
        for (int t = 0; t < NTILE; ++t) {
            bfrag[t] = Bp[((size_t)c * NTILE + t) * 32 + lane];
        }

        // Stage 128 tokens x 32 dims: fp32 -> bf16 into LDS (coalesced f4 loads)
#pragma unroll
        for (int i = 0; i < 4; ++i) {
            int idx = tid + i * 256;              // 0..1023
            int tok = idx >> 3;                   // 0..127
            int seg = idx & 7;                    // 8 segments of 4 floats
            float4 f = *(const float4*)(x + (size_t)(base + tok) * INPUT_DIM
                                          + kc + seg * 4);
            __bf16* p = &xs[buf][tok][seg * 4];
            p[0] = (__bf16)f.x; p[1] = (__bf16)f.y;
            p[2] = (__bf16)f.z; p[3] = (__bf16)f.w;
        }
        // Hint the chunk-after-next into cache (global_prefetch_b8)
        if (c + 2 < NCHUNK) {
            __builtin_prefetch(x + (size_t)(base + (tid >> 3)) * INPUT_DIM
                                 + (kc + 2 * KC) + (tid & 7) * 4, 0, 1);
        }
        __syncthreads();   // single barrier per iter; double-buffered xs

        v8bf alo = *(const v8bf*)&xs[buf][row][koff];
        v8bf ahi = *(const v8bf*)&xs[buf][row][koff + 8];
        v16bf a;
#pragma unroll
        for (int i = 0; i < 8; ++i) { a[i] = alo[i]; a[i + 8] = ahi[i]; }

#pragma unroll
        for (int t = 0; t < NTILE; ++t) {
            acc[t] = __builtin_amdgcn_wmma_f32_16x16x32_bf16(
                false, a, false, bfrag[t], (short)0, acc[t], false, false);
        }
    }

    // Scatter accumulators to LDS score tile.
    // C/D layout: VGPR r -> (M=r, N=lane) for lanes 0-15, (M=r+8, N=lane-16)
    // for lanes 16-31.
    const int nlo = lane & 15;
    const int hi8 = (lane >> 4) * 8;
#pragma unroll
    for (int t = 0; t < NTILE; ++t) {
#pragma unroll
        for (int r = 0; r < 8; ++r) {
            sc[wave * 16 + hi8 + r][t * 16 + nlo] = acc[t][r] + bsh[t * 16 + nlo];
        }
    }
    __syncthreads();

    // One thread per token: top-2 (first-occurrence ties, matching lax.top_k)
    // + softmax values at the top-2 experts.
    if (tid < TOK_TILE) {
        float v0 = -3.402823466e38f, v1 = -3.402823466e38f;
        int   i0 = 0, i1 = 0;
#pragma unroll 8
        for (int e = 0; e < NUM_EXPERTS; ++e) {
            float s = sc[tid][e];
            if (s > v0)      { v1 = v0; i1 = i0; v0 = s; i0 = e; }
            else if (s > v1) { v1 = s; i1 = e; }
        }
        float den = 0.f;
#pragma unroll 8
        for (int e = 0; e < NUM_EXPERTS; ++e) den += __expf(sc[tid][e] - v0);
        float inv = 1.0f / den;
        float p0  = inv;                       // exp(v0 - v0) / den
        float p1  = __expf(v1 - v0) * inv;

        int g = base + tid;
        float* idxo = out;                          // output 0: indices
        float* valo = out + NUM_TOKENS * TOP_K;     // output 1: values
        idxo[g * 2]     = (float)i0;
        idxo[g * 2 + 1] = (float)i1;
        valo[g * 2]     = p0;
        valo[g * 2 + 1] = p1;
    }
}

extern "C" void kernel_launch(void* const* d_in, const int* in_sizes, int n_in,
                              void* d_out, int out_size, void* d_ws, size_t ws_size,
                              hipStream_t stream) {
    const float* x    = (const float*)d_in[0];   // [16384, 4096] fp32
    const float* W    = (const float*)d_in[1];   // [64, 4096]    fp32
    const float* bias = (const float*)d_in[2];   // [64]          fp32

    __bf16* Bpack = (__bf16*)d_ws;               // 128*4*32*16 bf16 = 512 KB

    // Pack W into WMMA-ready bf16 B fragments (262144 elements).
    topk_gating_pack_w<<<(NCHUNK * NTILE * 32 * 16) / 256, 256, 0, stream>>>(W, Bpack);

    // Fused GEMM + softmax + top-2.
    topk_gating_gemm<<<NUM_TOKENS / TOK_TILE, 256, 0, stream>>>(
        x, Bpack, bias, (float*)d_out);
}
